// MultiHeadAttention_5772436045980
// MI455X (gfx1250) — matrix operand order repaired and checked by reading, not compiled, not run on hardware
//
#include <hip/hip_runtime.h>

// ---------------------------------------------------------------------------
// MHA forward for MI455X (gfx1250, wave32): bf16 WMMA w/ f32 accumulation.
// Pipeline: [QKV gemm + head scatter] -> [flash attention] -> [out gemm].
// GEMMs use double-buffered LDS staging (one barrier per K-step, global
// loads for tile k+1 issued before the WMMA chain for tile k).
// ---------------------------------------------------------------------------

#define D_MODEL 1024
#define N_HEADS 16
#define HEAD_DIM 64
#define SEQ 2048
#define BATCH 2
#define BT (BATCH * SEQ)      // 4096 rows of tokens
#define BH (BATCH * N_HEADS)  // 32 (batch, head) pairs

typedef __attribute__((ext_vector_type(16))) __bf16   bf16x16;
typedef __attribute__((ext_vector_type(8)))  float    floatx8;
typedef __attribute__((ext_vector_type(4)))  float    floatx4;
typedef __attribute__((ext_vector_type(4)))  unsigned uintx4;

union Frag {            // one 16x32 bf16 A fragment or 32x16 bf16 B fragment
  bf16x16 v;
  uintx4  q[2];
};

__device__ __forceinline__ floatx8 wmma_bf16(const Frag& a, const Frag& b, floatx8 c) {
  return __builtin_amdgcn_wmma_f32_16x16x32_bf16(false, a.v, false, b.v,
                                                 (short)0, c, false, false);
}

__device__ __forceinline__ unsigned pack2(float a, float b) {
  union { __bf16 h[2]; unsigned u; } t;
  t.h[0] = (__bf16)a; t.h[1] = (__bf16)b;
  return t.u;
}

__device__ __forceinline__ uintx4 pack8(const floatx4& a, const floatx4& b) {
  uintx4 r = { pack2(a[0], a[1]), pack2(a[2], a[3]),
               pack2(b[0], b[1]), pack2(b[2], b[3]) };
  return r;
}

// ---------------------------------------------------------------------------
// Kernel 1: qkv = x @ W_qkv + b_qkv, scattered into per-head Q / K^T / V bf16.
// Block: 256 thr (8 waves), tile 128(M) x 128(N), K-step 32, double-buffered.
// Wave (wm,wn) owns a 32x64 accumulator = 2x4 WMMA tiles -> 8 wmma / K-step.
// ---------------------------------------------------------------------------
__global__ __launch_bounds__(256) void qkv_gemm_kernel(
    const float* __restrict__ x, const float* __restrict__ W,
    const float* __restrict__ bias,
    __bf16* __restrict__ qb, __bf16* __restrict__ ktb, __bf16* __restrict__ vb) {
  constexpr int LDA = 40, LDB = 136;  // padded LDS strides (bf16 elements)
  __shared__ alignas(16) __bf16 As[2][128 * LDA];
  __shared__ alignas(16) __bf16 Bs[2][32 * LDB];
  const int tid = threadIdx.x, lane = tid & 31, wv = tid >> 5;
  const int wm = wv & 3, wn = wv >> 2;        // 4 x 2 wave grid
  const int g = lane >> 4, l16 = lane & 15;
  const int M0 = blockIdx.y * 128, N0 = blockIdx.x * 128;
  const int arow = tid >> 1, acol = (tid & 1) * 16;  // A stage: 16 f32/thread
  const int brow = tid >> 3, bcol = (tid & 7) * 16;  // B stage: 16 f32/thread

  const float* aptr = x + (size_t)(M0 + arow) * D_MODEL + acol;
  const float* bptr = W + (size_t)brow * (3 * D_MODEL) + N0 + bcol;

  floatx8 acc[2][4] = {};
  floatx4 ar[4], br[4];

  auto gload = [&](int k0) {
#pragma unroll
    for (int i = 0; i < 4; ++i) ar[i] = ((const floatx4*)(aptr + k0))[i];
#pragma unroll
    for (int i = 0; i < 4; ++i)
      br[i] = ((const floatx4*)(bptr + (size_t)k0 * (3 * D_MODEL)))[i];
  };
  auto stage = [&](int buf) {
    *(uintx4*)&As[buf][arow * LDA + acol]     = pack8(ar[0], ar[1]);
    *(uintx4*)&As[buf][arow * LDA + acol + 8] = pack8(ar[2], ar[3]);
    *(uintx4*)&Bs[buf][brow * LDB + bcol]     = pack8(br[0], br[1]);
    *(uintx4*)&Bs[buf][brow * LDB + bcol + 8] = pack8(br[2], br[3]);
  };
  auto compute = [&](int buf) {
    Frag af[2], bf[4];
#pragma unroll
    for (int mi = 0; mi < 2; ++mi) {  // A 16x32: K runs [g*8,+8) & [16+g*8,+8)
      const __bf16* base = &As[buf][(wm * 32 + mi * 16 + l16) * LDA + g * 8];
      af[mi].q[0] = *(const uintx4*)base;
      af[mi].q[1] = *(const uintx4*)(base + 16);
    }
#pragma unroll
    for (int ni = 0; ni < 4; ++ni) {  // B 32x16: N = l16, K runs [g*8,+8) & [16+g*8,+8)
      const __bf16* base = &Bs[buf][wn * 64 + ni * 16 + l16];
#pragma unroll
      for (int s = 0; s < 16; ++s) bf[ni].v[s] = base[((s & 8) * 2 + g * 8 + (s & 7)) * LDB];
    }
#pragma unroll
    for (int mi = 0; mi < 2; ++mi)
#pragma unroll
      for (int ni = 0; ni < 4; ++ni)
        acc[mi][ni] = wmma_bf16(af[mi], bf[ni], acc[mi][ni]);
  };

  gload(0);
  stage(0);
  __syncthreads();
  int buf = 0;
  for (int k0 = 32; k0 < D_MODEL; k0 += 32) {
    gload(k0);        // overlap with wmma chain below
    compute(buf);
    stage(buf ^ 1);   // other buffer: safe, prior barrier cleared its readers
    __syncthreads();
    buf ^= 1;
  }
  compute(buf);

  // Epilogue: C/D layout -> scatter into Q [bh][t][d], K^T [bh][d][t], V [bh][t][d]
#pragma unroll
  for (int mi = 0; mi < 2; ++mi)
#pragma unroll
    for (int ni = 0; ni < 4; ++ni)
#pragma unroll
      for (int r = 0; r < 8; ++r) {
        int m = M0 + wm * 32 + mi * 16 + r + g * 8;
        int n = N0 + wn * 64 + ni * 16 + l16;
        __bf16 hv = (__bf16)(acc[mi][ni][r] + bias[n]);
        int b = m >> 11, t = m & (SEQ - 1);
        int which = n >> 10, hn = n & 1023, h = hn >> 6, d = hn & 63;
        int bh = b * N_HEADS + h;
        if (which == 0)      qb [((size_t)bh * SEQ + t) * HEAD_DIM + d] = hv;
        else if (which == 1) ktb[((size_t)bh * HEAD_DIM + d) * SEQ + t] = hv;
        else                 vb [((size_t)bh * SEQ + t) * HEAD_DIM + d] = hv;
      }
}

// ---------------------------------------------------------------------------
// Kernel 2: flash attention. Block = 128 thr (4 waves); each wave owns 16 q
// rows of one (b,h) and streams key blocks of 32 with online softmax.
// Per key block: 4 WMMA for S = Q K^T, 4 WMMA for O += P V. No barriers.
// ---------------------------------------------------------------------------
__global__ __launch_bounds__(128) void attn_kernel(
    const __bf16* __restrict__ qb, const __bf16* __restrict__ ktb,
    const __bf16* __restrict__ vb, __bf16* __restrict__ ob) {
  constexpr int LDP = 40;
  __shared__ alignas(16) __bf16 Ps[4][16 * LDP];  // per-wave P transpose scratch
  const int tid = threadIdx.x, lane = tid & 31, wv = tid >> 5;
  const int g = lane >> 4, l16 = lane & 15;
  const int bh = blockIdx.x;               // 0..31
  const int b = bh >> 4, h = bh & 15;
  const int qrow0 = blockIdx.y * 64 + wv * 16;

  const __bf16* Qbase  = qb  + ((size_t)bh * SEQ + qrow0) * HEAD_DIM;
  const __bf16* KTbase = ktb + (size_t)bh * HEAD_DIM * SEQ;
  const __bf16* Vbase  = vb  + (size_t)bh * SEQ * HEAD_DIM;

  // Q fragments (A 16x32) for d-chunks 0 and 1, resident for the whole loop.
  Frag qf[2];
#pragma unroll
  for (int c = 0; c < 2; ++c) {
    const __bf16* p = Qbase + (size_t)l16 * HEAD_DIM + c * 32 + g * 8;
    qf[c].q[0] = *(const uintx4*)p;
    qf[c].q[1] = *(const uintx4*)(p + 16);
  }

  floatx8 oacc[4] = {};       // O tile 16x64 = 4 C tiles
  float mrow[8], lrow[8];
#pragma unroll
  for (int r = 0; r < 8; ++r) { mrow[r] = -1e30f; lrow[r] = 0.f; }

  __bf16* Pw = &Ps[wv][0];

  for (int kb = 0; kb < SEQ; kb += 32) {
    // S = Q K^T; two 16-key C tiles, K-dim = 64 (2 chained wmma each)
    floatx8 sacc[2];
#pragma unroll
    for (int t2 = 0; t2 < 2; ++t2) {
      Frag kf0, kf1;  // B: N = key (l16), K = d in table order, gathered from K^T
      const __bf16* p0 = KTbase + kb + t2 * 16 + l16;
      const __bf16* p1 = p0 + (size_t)32 * SEQ;
#pragma unroll
      for (int s = 0; s < 16; ++s) {
        size_t kd = (size_t)((s & 8) * 2 + g * 8 + (s & 7)) * SEQ;
        kf0.v[s] = p0[kd]; kf1.v[s] = p1[kd];
      }
      floatx8 s = {};
      s = wmma_bf16(qf[0], kf0, s);
      s = wmma_bf16(qf[1], kf1, s);
      sacc[t2] = s;
    }

    // Online softmax per row (row = qrow0 + r + 8g; 16 keys live across lanes)
#pragma unroll
    for (int r = 0; r < 8; ++r) {
      float s0 = sacc[0][r] * 0.125f;   // 1/sqrt(64)
      float s1 = sacc[1][r] * 0.125f;
      float mx = fmaxf(s0, s1);
      mx = fmaxf(mx, __shfl_xor(mx, 1, 16));
      mx = fmaxf(mx, __shfl_xor(mx, 2, 16));
      mx = fmaxf(mx, __shfl_xor(mx, 4, 16));
      mx = fmaxf(mx, __shfl_xor(mx, 8, 16));
      float mnew  = fmaxf(mrow[r], mx);
      float alpha = __expf(mrow[r] - mnew);
      float p0 = __expf(s0 - mnew);
      float p1 = __expf(s1 - mnew);
      float ps = p0 + p1;
      ps += __shfl_xor(ps, 1, 16);
      ps += __shfl_xor(ps, 2, 16);
      ps += __shfl_xor(ps, 4, 16);
      ps += __shfl_xor(ps, 8, 16);
      lrow[r] = lrow[r] * alpha + ps;
      mrow[r] = mnew;
#pragma unroll
      for (int ni = 0; ni < 4; ++ni) oacc[ni][r] *= alpha;
      // park P (C layout) in LDS so it can be regathered in A layout
      Pw[(r + 8 * g) * LDP + l16]      = (__bf16)p0;
      Pw[(r + 8 * g) * LDP + 16 + l16] = (__bf16)p1;
    }

    Frag pf;  // P as A 16x32 fragment (wave-private LDS; ds ops are in-order)
    {
      const __bf16* p = Pw + l16 * LDP + g * 8;
      pf.q[0] = *(const uintx4*)p;
      pf.q[1] = *(const uintx4*)(p + 16);
    }

    // O += P V : B N = d (l16), K = key offset in table order
#pragma unroll
    for (int ni = 0; ni < 4; ++ni) {
      Frag vf;
      const __bf16* p = Vbase + (size_t)kb * HEAD_DIM + ni * 16 + l16;
#pragma unroll
      for (int s = 0; s < 16; ++s) vf.v[s] = p[(size_t)((s & 8) * 2 + g * 8 + (s & 7)) * HEAD_DIM];
      oacc[ni] = wmma_bf16(pf, vf, oacc[ni]);
    }
  }

  // Normalize and write head-concat layout [B][T][C]
#pragma unroll
  for (int r = 0; r < 8; ++r) {
    float inv = 1.0f / lrow[r];
    int t = qrow0 + r + 8 * g;
    __bf16* orow = ob + ((size_t)b * SEQ + t) * D_MODEL + h * HEAD_DIM;
#pragma unroll
    for (int ni = 0; ni < 4; ++ni)
      orow[ni * 16 + l16] = (__bf16)(oacc[ni][r] * inv);
  }
}

// ---------------------------------------------------------------------------
// Kernel 3: out = attn_out(bf16) @ W_out + b_out -> f32.
// Block: 256 thr, tile 128(M) x 64(N), K-step 32, double-buffered.
// ---------------------------------------------------------------------------
__global__ __launch_bounds__(256) void out_gemm_kernel(
    const __bf16* __restrict__ A, const float* __restrict__ W,
    const float* __restrict__ bias, float* __restrict__ out) {
  constexpr int LDA = 40, LDB = 72;
  __shared__ alignas(16) __bf16 As[2][128 * LDA];
  __shared__ alignas(16) __bf16 Bs[2][32 * LDB];
  const int tid = threadIdx.x, lane = tid & 31, wv = tid >> 5;
  const int wm = wv & 3, wn = wv >> 2;
  const int g = lane >> 4, l16 = lane & 15;
  const int M0 = blockIdx.y * 128, N0 = blockIdx.x * 64;
  const int arow = tid >> 1, acol = (tid & 1) * 16;  // A: 16 bf16/thread
  const int brow = tid >> 3, bcol = (tid & 7) * 8;   // B: 8 f32/thread

  const __bf16* aptr = A + (size_t)(M0 + arow) * D_MODEL + acol;
  const float*  bptr = W + (size_t)brow * D_MODEL + N0 + bcol;

  floatx8 acc[2][2] = {};
  uintx4 ar0, ar1;
  floatx4 br0, br1;

  auto gload = [&](int k0) {
    const uintx4* ap = (const uintx4*)(aptr + k0);
    ar0 = ap[0]; ar1 = ap[1];
    const floatx4* bp = (const floatx4*)(bptr + (size_t)k0 * D_MODEL);
    br0 = bp[0]; br1 = bp[1];
  };
  auto stage = [&](int buf) {
    *(uintx4*)&As[buf][arow * LDA + acol]     = ar0;
    *(uintx4*)&As[buf][arow * LDA + acol + 8] = ar1;
    *(uintx4*)&Bs[buf][brow * LDB + bcol]     = pack8(br0, br1);
  };
  auto compute = [&](int buf) {
    Frag af[2], bf[2];
#pragma unroll
    for (int mi = 0; mi < 2; ++mi) {
      const __bf16* base = &As[buf][(wm * 32 + mi * 16 + l16) * LDA + g * 8];
      af[mi].q[0] = *(const uintx4*)base;
      af[mi].q[1] = *(const uintx4*)(base + 16);
    }
#pragma unroll
    for (int ni = 0; ni < 2; ++ni) {
      const __bf16* base = &Bs[buf][wn * 32 + ni * 16 + l16];
#pragma unroll
      for (int s = 0; s < 16; ++s) bf[ni].v[s] = base[((s & 8) * 2 + g * 8 + (s & 7)) * LDB];
    }
#pragma unroll
    for (int mi = 0; mi < 2; ++mi)
#pragma unroll
      for (int ni = 0; ni < 2; ++ni)
        acc[mi][ni] = wmma_bf16(af[mi], bf[ni], acc[mi][ni]);
  };

  gload(0);
  stage(0);
  __syncthreads();
  int buf = 0;
  for (int k0 = 32; k0 < D_MODEL; k0 += 32) {
    gload(k0);
    compute(buf);
    stage(buf ^ 1);
    __syncthreads();
    buf ^= 1;
  }
  compute(buf);

#pragma unroll
  for (int mi = 0; mi < 2; ++mi)
#pragma unroll
    for (int ni = 0; ni < 2; ++ni)
#pragma unroll
      for (int r = 0; r < 8; ++r) {
        int m = M0 + wm * 32 + mi * 16 + r + g * 8;
        int n = N0 + wn * 32 + ni * 16 + l16;
        out[(size_t)m * D_MODEL + n] = acc[mi][ni][r] + bias[n];
      }
}

// ---------------------------------------------------------------------------
extern "C" void kernel_launch(void* const* d_in, const int* in_sizes, int n_in,
                              void* d_out, int out_size, void* d_ws, size_t ws_size,
                              hipStream_t stream) {
  const float* x     = (const float*)d_in[0];
  const float* W_qkv = (const float*)d_in[1];
  const float* b_qkv = (const float*)d_in[2];
  const float* W_out = (const float*)d_in[3];
  const float* b_out = (const float*)d_in[4];
  float* out = (float*)d_out;

  // Workspace: 4 x (BH*SEQ*HEAD_DIM) bf16 buffers = 32 MiB total.
  const size_t HEADSZ = (size_t)BH * SEQ * HEAD_DIM;  // 4M elements
  __bf16* qb  = (__bf16*)d_ws;
  __bf16* ktb = qb  + HEADSZ;
  __bf16* vb  = ktb + HEADSZ;
  __bf16* ob  = vb  + HEADSZ;

  // 1) fused QKV projection + per-head scatter (K stored pre-transposed)
  qkv_gemm_kernel<<<dim3((3 * D_MODEL) / 128, BT / 128), 256, 0, stream>>>(
      x, W_qkv, b_qkv, qb, ktb, vb);
  // 2) flash attention per (b,h) x 64-row q block
  attn_kernel<<<dim3(BH, SEQ / 64), 128, 0, stream>>>(qb, ktb, vb, ob);
  // 3) output projection
  out_gemm_kernel<<<dim3(D_MODEL / 64, BT / 128), 256, 0, stream>>>(
      ob, W_out, b_out, out);
}
